// MS_SSA_Conv_79216376807572
// MI455X (gfx1250) — matrix operand order repaired and checked by reading, not compiled, not run on hardware
//
#include <hip/hip_runtime.h>

#define T_ 4
#define B_ 16
#define C_ 512
#define V_ 1024
#define H_ 8
#define DH_ 64
#define BN_EPS 1e-5f

typedef __attribute__((ext_vector_type(16))) _Float16 v16h;
typedef __attribute__((ext_vector_type(8)))  _Float16 v8h;
typedef __attribute__((ext_vector_type(8)))  float    v8f;
typedef __attribute__((ext_vector_type(4)))  float    v4f;

// ---------------- weight fp32 -> f16 ----------------
__global__ __launch_bounds__(256)
void cvt_weights(const float* __restrict__ qw, const float* __restrict__ kw,
                 const float* __restrict__ vw, const float* __restrict__ pw,
                 _Float16* __restrict__ wh) {
  int i = blockIdx.x * blockDim.x + threadIdx.x;   // over C_*C_
  wh[i]               = (_Float16)qw[i];
  wh[C_ * C_ + i]     = (_Float16)kw[i];
  wh[2 * C_ * C_ + i] = (_Float16)vw[i];
  wh[3 * C_ * C_ + i] = (_Float16)pw[i];
}

// ---------------- BN scale/shift precompute (p folds bias) ----------------
__global__ __launch_bounds__(512)
void bn_prep(const float* __restrict__ qg, const float* __restrict__ qb,
             const float* __restrict__ qm, const float* __restrict__ qv,
             const float* __restrict__ kg, const float* __restrict__ kb,
             const float* __restrict__ km, const float* __restrict__ kvr,
             const float* __restrict__ vg, const float* __restrict__ vb,
             const float* __restrict__ vm, const float* __restrict__ vv,
             const float* __restrict__ pg, const float* __restrict__ pb,
             const float* __restrict__ pm, const float* __restrict__ pv,
             const float* __restrict__ pbias, float* __restrict__ out) {
  int c = threadIdx.x;  // 0..511
  float i0 = qg[c] * rsqrtf(qv[c] + BN_EPS);
  out[c] = i0;          out[512 + c]  = qb[c] - qm[c] * i0;
  float i1 = kg[c] * rsqrtf(kvr[c] + BN_EPS);
  out[1024 + c] = i1;   out[1536 + c] = kb[c] - km[c] * i1;
  float i2 = vg[c] * rsqrtf(vv[c] + BN_EPS);
  out[2048 + c] = i2;   out[2560 + c] = vb[c] - vm[c] * i2;
  float i3 = pg[c] * rsqrtf(pv[c] + BN_EPS);
  out[3072 + c] = i3;   out[3584 + c] = pb[c] - pm[c] * i3 + pbias[c] * i3;
}

// ---------- shortcut LIF on x -> transposed f16 spikes (LDS transpose) ----------
__global__ __launch_bounds__(256)
void lif_x(const float* __restrict__ x, _Float16* __restrict__ xsT) {
  __shared__ _Float16 sp[T_][32][33];
  const int b   = blockIdx.z;
  const int c0  = blockIdx.y * 32;
  const int v0  = blockIdx.x * 32;
  const int tid = threadIdx.x;
  const int vv  = tid & 31;
  const int cb  = tid >> 5;   // 0..7
#pragma unroll
  for (int j = 0; j < 4; ++j) {
    const int cc = cb + j * 8;
    float vmem = 0.f;
#pragma unroll
    for (int t = 0; t < T_; ++t) {
      float xv = x[(((size_t)(t * B_ + b) * C_) + c0 + cc) * V_ + v0 + vv];
      vmem = 0.5f * (vmem + xv);            // v += (x - v)/tau, tau=2
      float s = (vmem >= 1.0f) ? 1.0f : 0.0f;
      vmem *= (1.0f - s);                   // hard reset
      sp[t][cc][vv] = (_Float16)s;
    }
  }
  __syncthreads();
  const int cw = tid & 31;
  const int vb = tid >> 5;
#pragma unroll
  for (int j = 0; j < 4; ++j) {
    const int vr = vb + j * 8;
#pragma unroll
    for (int t = 0; t < T_; ++t)
      xsT[((size_t)(t * B_ + b) * V_ + v0 + vr) * C_ + c0 + cw] = sp[t][vr][cw];
  }
}

// ---------------- WMMA GEMM + BN (+LIF), LDS double-buffered ----------------
// Wave tile 32M x 32N x all-4T; workgroup 128M x 64N (waves 4x2).
// Each B fragment feeds two WMMAs (reused across the two A subtiles) ->
// halves LDS read traffic per WMMA vs a 16Mx64N arrangement.
// MODE 0: q/k branch -> f16 transposed spikes (dstT), packed v8h stores
// MODE 1: v branch   -> fp32 spikes in (T,B,H,V,Dh) layout (dstF = d_out part2)
// MODE 2: projection -> B scaled by kv spikes at staging; out = bn(gemm)+x (dstF)
template <int MODE>
__global__ __launch_bounds__(256)
void gemm_lif(const _Float16* __restrict__ Wh,
              const _Float16* __restrict__ Bsrc,
              const _Float16* __restrict__ kvs,
              const float* __restrict__ bninv,
              const float* __restrict__ bnshift,
              const float* __restrict__ xin,
              _Float16* __restrict__ dstT,
              float* __restrict__ dstF) {
  constexpr int LROW = 40;  // halves per LDS row: 80B stride, 16B aligned, bank-spread
  __shared__ _Float16 ldsB[2][T_ * 64 * LROW];   // 2 x 20KB

  const int tid  = threadIdx.x;
  const int lane = tid & 31;
  const int wv   = tid >> 5;     // 0..7
  const int wm   = wv & 3;       // M subtile (4)
  const int wn   = wv >> 2;      // N subtile (2)
  const int hs   = lane >> 4;
  const int lm   = lane & 15;
  const int b    = blockIdx.z;
  const int m0   = blockIdx.y * 128 + wm * 32;
  const int n0   = blockIdx.x * 64;          // workgroup N base (staging)
  const int n0w  = n0 + wn * 32;             // wave N base

  // staging role: one 64B row (32 halves of K) per thread
  const int ts = tid >> 6;   // 0..3 timestep
  const int ns = tid & 63;   // 0..63 column
  const _Float16* gB = Bsrc + (size_t)(ts * B_ + b) * V_ * C_ + (size_t)(n0 + ns) * C_;
  const _Float16* gK = (MODE == 2) ? (kvs + (size_t)(ts * B_ + b) * C_) : (const _Float16*)nullptr;
  _Float16* lrow[2] = {&ldsB[0][(ts * 64 + ns) * LROW], &ldsB[1][(ts * 64 + ns) * LROW]};

  const _Float16* arow0 = Wh + (size_t)(m0 + lm) * C_ + hs * 16;       // rows m0..m0+15
  const _Float16* arow1 = arow0 + (size_t)16 * C_;                     // rows m0+16..+31

  v8f zero = {0.f, 0.f, 0.f, 0.f, 0.f, 0.f, 0.f, 0.f};
  v8f acc[T_][2][2];   // [t][mt][nt]
#pragma unroll
  for (int t = 0; t < T_; ++t)
#pragma unroll
    for (int mt = 0; mt < 2; ++mt)
#pragma unroll
      for (int nt = 0; nt < 2; ++nt) acc[t][mt][nt] = zero;

  // ---- prologue: stage k=0, load A(k=0) ----
  {
    v16h s0 = *(const v16h*)(gB + 0);
    v16h s1 = *(const v16h*)(gB + 16);
    if (MODE == 2) {
      s0 *= *(const v16h*)(gK + 0);
      s1 *= *(const v16h*)(gK + 16);
    }
    *(v16h*)(lrow[0] + 0)  = s0;
    *(v16h*)(lrow[0] + 16) = s1;
  }
  v8h alo0 = *(const v8h*)(arow0),      ahi0 = *(const v8h*)(arow0 + 8);
  v8h alo1 = *(const v8h*)(arow1),      ahi1 = *(const v8h*)(arow1 + 8);
  __syncthreads();

  int cur = 0;
  for (int k0 = 0; k0 < C_; k0 += 32) {
    const bool more = (k0 + 32 < C_);
    v16h ns0, ns1;
    v8h  nalo0, nahi0, nalo1, nahi1;
    if (more) {  // issue next-tile global loads early; consumed after compute
      ns0 = *(const v16h*)(gB + k0 + 32);
      ns1 = *(const v16h*)(gB + k0 + 48);
      if (MODE == 2) {
        ns0 *= *(const v16h*)(gK + k0 + 32);
        ns1 *= *(const v16h*)(gK + k0 + 48);
      }
      nalo0 = *(const v8h*)(arow0 + k0 + 32);
      nahi0 = *(const v8h*)(arow0 + k0 + 40);
      nalo1 = *(const v8h*)(arow1 + k0 + 32);
      nahi1 = *(const v8h*)(arow1 + k0 + 40);
    }
    v16h a0 = __builtin_shufflevector(alo0, ahi0, 0, 1, 2, 3, 4, 5, 6, 7,
                                      8, 9, 10, 11, 12, 13, 14, 15);
    v16h a1 = __builtin_shufflevector(alo1, ahi1, 0, 1, 2, 3, 4, 5, 6, 7,
                                      8, 9, 10, 11, 12, 13, 14, 15);
    const _Float16* lbase = &ldsB[cur][0] + hs * 16;
#pragma unroll
    for (int t = 0; t < T_; ++t) {
      v16h bf0 = *(const v16h*)(lbase + (t * 64 + wn * 32 + lm) * LROW);
      v16h bf1 = *(const v16h*)(lbase + (t * 64 + wn * 32 + 16 + lm) * LROW);
      acc[t][0][0] = __builtin_amdgcn_wmma_f32_16x16x32_f16(
          false, a0, false, bf0, (short)0, acc[t][0][0], false, false);
      acc[t][1][0] = __builtin_amdgcn_wmma_f32_16x16x32_f16(
          false, a1, false, bf0, (short)0, acc[t][1][0], false, false);
      acc[t][0][1] = __builtin_amdgcn_wmma_f32_16x16x32_f16(
          false, a0, false, bf1, (short)0, acc[t][0][1], false, false);
      acc[t][1][1] = __builtin_amdgcn_wmma_f32_16x16x32_f16(
          false, a1, false, bf1, (short)0, acc[t][1][1], false, false);
    }
    if (more) {
      _Float16* lw = lrow[cur ^ 1];
      *(v16h*)(lw + 0)  = ns0;
      *(v16h*)(lw + 16) = ns1;
      alo0 = nalo0; ahi0 = nahi0; alo1 = nalo1; ahi1 = nahi1;
      __syncthreads();
      cur ^= 1;
    }
  }

  // ---- epilogue ----
#pragma unroll
  for (int mt = 0; mt < 2; ++mt) {
    const int mstart = m0 + mt * 16 + hs * 8;   // 8 contiguous channels per lane
    float ivv[8], shv[8];
#pragma unroll
    for (int r = 0; r < 8; ++r) {
      ivv[r] = bninv[mstart + r];
      shv[r] = bnshift[mstart + r];
    }
#pragma unroll
    for (int nt = 0; nt < 2; ++nt) {
      const int n = n0w + nt * 16 + lm;
      if (MODE == 2) {
#pragma unroll
        for (int t = 0; t < T_; ++t)
#pragma unroll
          for (int r = 0; r < 8; ++r) {
            size_t idx = (((size_t)(t * B_ + b) * C_) + mstart + r) * V_ + n;
            dstF[idx] = acc[t][mt][nt][r] * ivv[r] + shv[r] + xin[idx];  // bn + identity
          }
      } else {
        float vmem[8] = {0.f, 0.f, 0.f, 0.f, 0.f, 0.f, 0.f, 0.f};
#pragma unroll
        for (int t = 0; t < T_; ++t) {
          v8h sp;
          v4f slo, shi;
#pragma unroll
          for (int r = 0; r < 8; ++r) {
            float y = acc[t][mt][nt][r] * ivv[r] + shv[r];
            vmem[r] = 0.5f * (vmem[r] + y);
            float s = (vmem[r] >= 1.0f) ? 1.0f : 0.0f;
            vmem[r] *= (1.0f - s);
            if (MODE == 0) sp[r] = (_Float16)s;
            else { if (r < 4) slo[r] = s; else shi[r - 4] = s; }
          }
          if (MODE == 0) {
            // 16B packed spike store: channels mstart..mstart+7 contiguous
            *(v8h*)(dstT + ((size_t)(t * B_ + b) * V_ + n) * C_ + mstart) = sp;
          } else {  // MODE 1: v output, (T,B,H,N,Dh) fp32; 2x16B stores
            const int h = mstart >> 6, d0 = mstart & 63;
            float* p = dstF + ((((size_t)(t * B_ + b) * H_) + h) * V_ + n) * DH_ + d0;
            *(v4f*)(p) = slo;
            *(v4f*)(p + 4) = shi;
          }
        }
      }
    }
  }
}

// ---------------- kv = sum_n k*v, then LIF(thr=0.5) over T ----------------
__global__ __launch_bounds__(512)
void kv_lif(const _Float16* __restrict__ kT, const float* __restrict__ vout,
            _Float16* __restrict__ kvs) {
  const int c = threadIdx.x;   // 0..511 (contiguous in kT innermost dim)
  const int b = blockIdx.x;    // 0..15
  const int h = c >> 6, d = c & 63;
  float s[T_] = {0.f, 0.f, 0.f, 0.f};
  for (int n = 0; n < V_; ++n) {
#pragma unroll
    for (int t = 0; t < T_; ++t) {
      float kk = (float)kT[((size_t)(t * B_ + b) * V_ + n) * C_ + c];
      float vv = vout[((((size_t)(t * B_ + b) * H_) + h) * V_ + n) * DH_ + d];
      s[t] += kk * vv;
    }
  }
  float vmem = 0.f;
#pragma unroll
  for (int t = 0; t < T_; ++t) {
    vmem = 0.5f * (vmem + s[t]);
    float sp = (vmem >= 0.5f) ? 1.0f : 0.0f;
    vmem *= (1.0f - sp);
    kvs[(size_t)(t * B_ + b) * C_ + c] = (_Float16)sp;
  }
}

// ---------------- workspace layout (bytes, all 256-aligned) ----------------
static constexpr size_t OFF_WH  = 0;                              // 4*C*C f16  = 2 MiB
static constexpr size_t OFF_BN  = 2097152;                        // 4096 f32
static constexpr size_t OFF_KVS = OFF_BN + 16384;                 // T*B*C f16
static constexpr size_t OFF_XST = OFF_KVS + 65536;                // T*B*V*C f16 = 64 MiB
static constexpr size_t SZ_SPK  = (size_t)T_ * B_ * V_ * C_ * 2;
static constexpr size_t OFF_QT  = OFF_XST + SZ_SPK;
static constexpr size_t OFF_KT  = OFF_QT + SZ_SPK;                // end ~194.1 MiB

extern "C" void kernel_launch(void* const* d_in, const int* in_sizes, int n_in,
                              void* d_out, int out_size, void* d_ws, size_t ws_size,
                              hipStream_t stream) {
  const float* x    = (const float*)d_in[0];
  const float* q_w  = (const float*)d_in[1];
  const float* q_g  = (const float*)d_in[2];
  const float* q_b  = (const float*)d_in[3];
  const float* q_m  = (const float*)d_in[4];
  const float* q_v  = (const float*)d_in[5];
  const float* k_w  = (const float*)d_in[6];
  const float* k_g  = (const float*)d_in[7];
  const float* k_b  = (const float*)d_in[8];
  const float* k_m  = (const float*)d_in[9];
  const float* k_v  = (const float*)d_in[10];
  const float* v_w  = (const float*)d_in[11];
  const float* v_g  = (const float*)d_in[12];
  const float* v_b  = (const float*)d_in[13];
  const float* v_m  = (const float*)d_in[14];
  const float* v_v  = (const float*)d_in[15];
  const float* p_w  = (const float*)d_in[16];
  const float* p_b  = (const float*)d_in[17];
  const float* p_g  = (const float*)d_in[18];
  const float* p_be = (const float*)d_in[19];
  const float* p_m  = (const float*)d_in[20];
  const float* p_v  = (const float*)d_in[21];

  char* ws = (char*)d_ws;
  _Float16* Wh  = (_Float16*)(ws + OFF_WH);
  float*    bnb = (float*)(ws + OFF_BN);
  _Float16* kvs = (_Float16*)(ws + OFF_KVS);
  _Float16* xsT = (_Float16*)(ws + OFF_XST);
  _Float16* qT  = (_Float16*)(ws + OFF_QT);
  _Float16* kT  = (_Float16*)(ws + OFF_KT);

  float* outp = (float*)d_out;                       // (T,B,C,V)
  float* vout = outp + (size_t)T_ * B_ * C_ * V_;    // (T,B,H,N,Dh)

  cvt_weights<<<(C_ * C_) / 256, 256, 0, stream>>>(q_w, k_w, v_w, p_w, Wh);
  bn_prep<<<1, 512, 0, stream>>>(q_g, q_b, q_m, q_v, k_g, k_b, k_m, k_v,
                                 v_g, v_b, v_m, v_v, p_g, p_be, p_m, p_v,
                                 p_b, bnb);
  {
    dim3 gl(V_ / 32, C_ / 32, B_);
    lif_x<<<gl, 256, 0, stream>>>(x, xsT);
  }

  dim3 g(V_ / 64, C_ / 128, B_), blk(256);
  gemm_lif<0><<<g, blk, 0, stream>>>(Wh,               xsT, nullptr, bnb,        bnb + 512,  nullptr, qT,      nullptr);
  gemm_lif<0><<<g, blk, 0, stream>>>(Wh + C_ * C_,     xsT, nullptr, bnb + 1024, bnb + 1536, nullptr, kT,      nullptr);
  gemm_lif<1><<<g, blk, 0, stream>>>(Wh + 2 * C_ * C_, xsT, nullptr, bnb + 2048, bnb + 2560, nullptr, nullptr, vout);
  kv_lif<<<B_, 512, 0, stream>>>(kT, vout, kvs);
  gemm_lif<2><<<g, blk, 0, stream>>>(Wh + 3 * C_ * C_, qT,  kvs,     bnb + 3072, bnb + 3584, x,       nullptr, outp);
}
